// SelfAttention_19971597926571
// MI455X (gfx1250) — matrix operand order repaired and checked by reading, not compiled, not run on hardware
//
#include <hip/hip_runtime.h>
#include <stdint.h>

typedef __attribute__((ext_vector_type(16))) _Float16     v16h;
typedef __attribute__((ext_vector_type(8)))  float        v8f;
typedef __attribute__((ext_vector_type(4)))  float        v4f;
typedef __attribute__((ext_vector_type(4)))  _Float16     v4h;
typedef __attribute__((ext_vector_type(4)))  unsigned int v4u;
typedef __attribute__((ext_vector_type(4)))  int          v4i;
typedef __attribute__((ext_vector_type(8)))  int          v8i;

#define SEQ_T 128
#define DIM_C 128
#define XS 136   // f16 LDS row stride (elems) -> 272B = 68 banks, kills 64-bank periodicity
#define SS 132   // f32 LDS row stride (elems) -> 528B

// ---- LDS layout (bytes); f16 panel = 128*136*2 = 34816 --------------------
#define OFF_QS   0
#define OFF_KS   34816
#define OFF_VT   69632         // v stored transposed [h][t]
#define OFF_XS   104448        // x panel, reused as softmax-weights panel
#define OFF_WQ   139264
#define OFF_WK   174080
#define OFF_WV   208896
#define OFF_SC   243712        // scores f32: 128*132*4 = 67584
#define OFF_RED  311296        // 256 f32 partial-max
#define OFF_RED2 312320        // 256 f32 partial-sum
#define LDS_BYTES 313344       // = 153 * 2048, fits 320KB WGP LDS

// ---- TDM availability guard ----------------------------------------------
#if defined(__has_builtin)
#if __has_builtin(__builtin_amdgcn_tensor_load_to_lds) && \
    __has_builtin(__builtin_amdgcn_s_wait_tensorcnt)
#define USE_TDM 1
#endif
#endif
#ifndef USE_TDM
#define USE_TDM 0
#endif

static __device__ __forceinline__ v8f wmma16(v16h a, v16h b, v8f c) {
  return __builtin_amdgcn_wmma_f32_16x16x32_f16(false, a, false, b, (short)0, c, false, false);
}

// A fragment (16-bit 16x32 MxK): lane L -> M = m0+(L&15); K = (L>>4)*16 + 2v (+1).
// One contiguous 32B run per lane (matches frag_b_row).
static __device__ __forceinline__ v16h frag_a(const _Float16* p, int m0, int k0, int lane) {
  const _Float16* row = p + (m0 + (lane & 15)) * XS + k0 + ((lane >> 4) << 4);
  v16h a;
#pragma unroll
  for (int v = 0; v < 4; ++v) {
    a[2 * v]         = row[2 * v];
    a[2 * v + 1]     = row[2 * v + 1];
    a[8 + 2 * v]     = row[8 + 2 * v];
    a[8 + 2 * v + 1] = row[8 + 2 * v + 1];
  }
  return a;
}

// B fragment (16-bit 32x16 KxN): lane L -> N = n0+(L&15); K = (L>>4)*16 + 2v (+1).
// p indexed p[n*XS + k] (row n supplies column n) -> one contiguous 32B run per lane.
static __device__ __forceinline__ v16h frag_b_row(const _Float16* p, int n0, int k0, int lane) {
  const _Float16* row = p + (n0 + (lane & 15)) * XS + k0 + ((lane >> 4) << 4);
  v16h b;
#pragma unroll
  for (int v = 0; v < 8; ++v) {
    b[2 * v]     = row[2 * v];
    b[2 * v + 1] = row[2 * v + 1];
  }
  return b;
}

#if USE_TDM
// TDM: DMA one 128-row x 256B f16 panel from global into LDS, hardware-padded
// to a 272B row stride (pad_interval: 64 DWORDs -> code 5; pad_amount: 4 DWORDs -> code 3).
// D# group0: count=1 | lds_addr | global_addr[56:0] | type=2.
// D# group1: data_size=4B(code 2), pad_enable, tensor_dim0=64dw, tensor_dim1=128,
//            tile_dim0=64dw, tile_dim1=128, tensor_dim0_stride=64dw.
// This toolchain's builtin takes 6 args: (g0, g1, g2, g3, g4, cpol); extra
// descriptor groups are zero (2D tile, no iteration/gather, not in a cluster).
static __device__ __forceinline__ void tdm_load_panel(const _Float16* gsrc, unsigned lds_off) {
  uint64_t ga = (uint64_t)(uintptr_t)gsrc;
  v4u g0 = { 1u,                                   // count=1 (valid user descriptor)
             lds_off,                              // lds_addr (bytes)
             (unsigned)(ga & 0xFFFFFFFFu),         // global_addr[31:0]
             (unsigned)((ga >> 32) & 0x01FFFFFFu) | (2u << 30) };  // [56:32] | type=2
  v8i g1 = { (int)((2u << 16) | (1u << 20) | (5u << 22) | (3u << 25)),
             (int)(64u << 16),                     // tensor_dim0[15:0] @ bits63:48
             (int)(128u << 16),                    // tensor_dim1[15:0] @ bits95:80
             (int)(64u << 16),                     // tile_dim0 @ bits127:112
             128,                                  // tile_dim1 @ bits143:128
             64,                                   // tensor_dim0_stride[31:0]
             0, 0 };
  v4i z4 = { 0, 0, 0, 0 };
  v8i z8 = { 0, 0, 0, 0, 0, 0, 0, 0 };
  __builtin_amdgcn_tensor_load_to_lds(g0, g1, z4, z4, z8, 0);
}
#endif

// One 16x128 output tile: D = A(16x128) * B(128x128); B given row-transposed.
static __device__ __forceinline__ void proj_tile(const _Float16* A, const _Float16* Brows,
                                                 _Float16* D, int m0, int lane, bool transposeD) {
  v16h a0 = frag_a(A, m0, 0, lane);
  v16h a1 = frag_a(A, m0, 32, lane);
  v16h a2 = frag_a(A, m0, 64, lane);
  v16h a3 = frag_a(A, m0, 96, lane);
  const int mrow = m0 + ((lane >> 4) << 3);
  for (int nt = 0; nt < 8; ++nt) {
    const int n0 = nt << 4;
    v8f acc = {};
    acc = wmma16(a0, frag_b_row(Brows, n0, 0, lane), acc);
    acc = wmma16(a1, frag_b_row(Brows, n0, 32, lane), acc);
    acc = wmma16(a2, frag_b_row(Brows, n0, 64, lane), acc);
    acc = wmma16(a3, frag_b_row(Brows, n0, 96, lane), acc);
    const int ncol = n0 + (lane & 15);
#pragma unroll
    for (int i = 0; i < 8; ++i) {
      _Float16 h = (_Float16)acc[i];
      if (!transposeD) D[(mrow + i) * XS + ncol] = h;
      else             D[ncol * XS + (mrow + i)] = h;
    }
  }
}

// Pre-kernel: convert Wq|Wk|Wv (f32, 3x16384) once into contiguous f16 in d_ws.
__global__ __launch_bounds__(256)
void wcvt_kernel(const float* __restrict__ Wq, const float* __restrict__ Wk,
                 const float* __restrict__ Wv, _Float16* __restrict__ dst) {
  const int i = blockIdx.x * 256 + threadIdx.x;        // 0..49151
  const float* s = (i < 16384) ? (Wq + i)
                 : (i < 32768) ? (Wk + (i - 16384))
                                : (Wv + (i - 32768));
  dst[i] = (_Float16)(*s);
}

__global__ __launch_bounds__(256)
void attn128_kernel(const float* __restrict__ x, const _Float16* __restrict__ wf16,
                    float* __restrict__ out) {
  extern __shared__ char smem[];
  _Float16* qs  = (_Float16*)(smem + OFF_QS);
  _Float16* ks  = (_Float16*)(smem + OFF_KS);
  _Float16* vt  = (_Float16*)(smem + OFF_VT);
  _Float16* xs  = (_Float16*)(smem + OFF_XS);
  _Float16* ws  = xs;
  _Float16* wmq = (_Float16*)(smem + OFF_WQ);
  _Float16* wmk = (_Float16*)(smem + OFF_WK);
  _Float16* wmv = (_Float16*)(smem + OFF_WV);
  float*    sc  = (float*)(smem + OFF_SC);
  float*    red  = (float*)(smem + OFF_RED);
  float*    red2 = (float*)(smem + OFF_RED2);

  const int b    = blockIdx.x;
  const int tid  = threadIdx.x;
  const int lane = tid & 31;
  const int wid  = tid >> 5;          // 8 waves, one 16-row tile each

  // --- stage weights (TDM DMA, overlapped with x staging) ----------------
#if USE_TDM
  if (tid < 32) {                     // wave 0 issues 3 tensor DMAs
    const unsigned base = (unsigned)(uintptr_t)smem;
    tdm_load_panel(wf16,         base + OFF_WQ);
    tdm_load_panel(wf16 + 16384, base + OFF_WK);
    tdm_load_panel(wf16 + 32768, base + OFF_WV);
  }
#else
  for (int c = tid; c < 3 * 2048; c += 256) {          // 16B chunks
    const int p = c >> 11, cc = c & 2047;
    const int r = cc >> 4, k16 = cc & 15;
    const uint4 v = ((const uint4*)(wf16 + p * 16384))[cc];
    *(uint4*)(smem + OFF_WQ + p * 34816 + r * 272 + k16 * 16) = v;
  }
#endif

  // --- stage x[b]: f32 global -> f16 LDS, float4 granularity -------------
  {
    const float* xb = x + (size_t)b * SEQ_T * DIM_C;
    for (int c = tid; c < 4096; c += 256) {            // 4096 float4 chunks
      const int r = c >> 5, col = (c & 31) << 2;
      v4f s = ((const v4f*)xb)[c];
      v4h d;
#pragma unroll
      for (int e = 0; e < 4; ++e) d[e] = (_Float16)s[e];
      *(v4h*)(xs + r * XS + col) = d;
    }
  }
#if USE_TDM
  if (tid < 32) __builtin_amdgcn_s_wait_tensorcnt(0);
#endif
  __syncthreads();

  // --- projections: each wave computes its 16 rows of q, k, v ------------
  const int m0 = wid << 4;
  proj_tile(xs, wmq, qs, m0, lane, false);
  proj_tile(xs, wmk, ks, m0, lane, false);
  proj_tile(xs, wmv, vt, m0, lane, true);              // v stored [h][t]
  __syncthreads();

  // --- scores = q . k^T, lower-triangle tiles only (causal) --------------
  {
    v16h a0 = frag_a(qs, m0, 0, lane);
    v16h a1 = frag_a(qs, m0, 32, lane);
    v16h a2 = frag_a(qs, m0, 64, lane);
    v16h a3 = frag_a(qs, m0, 96, lane);
    const int mrow = m0 + ((lane >> 4) << 3);
    for (int nt = 0; nt <= wid; ++nt) {                // wave-uniform bound
      const int n0 = nt << 4;
      v8f acc = {};
      acc = wmma16(a0, frag_b_row(ks, n0, 0, lane), acc);
      acc = wmma16(a1, frag_b_row(ks, n0, 32, lane), acc);
      acc = wmma16(a2, frag_b_row(ks, n0, 64, lane), acc);
      acc = wmma16(a3, frag_b_row(ks, n0, 96, lane), acc);
      const int ncol = n0 + (lane & 15);
#pragma unroll
      for (int i = 0; i < 8; ++i) sc[(mrow + i) * SS + ncol] = acc[i];
    }
  }
  __syncthreads();

  // --- causal softmax: 2 threads per row, vectorized ---------------------
  {
    const float scale = 0.08838834764831845f;          // 128^-0.5
    const int t = tid >> 1, h = tid & 1;
    const int jb0 = h << 6;                            // own half: 64 cols
    v4f* sp = (v4f*)(sc + t * SS + jb0);
    float mx = -3.0e38f;
#pragma unroll 4
    for (int c = 0; c < 16; ++c) {
      v4f v = sp[c];
#pragma unroll
      for (int e = 0; e < 4; ++e) {
        const int j = jb0 + c * 4 + e;
        if (j <= t) mx = fmaxf(mx, v[e] * scale);
      }
    }
    red[tid] = mx;
    __syncthreads();
    const float m = fmaxf(red[2 * t], red[2 * t + 1]);
    float sum = 0.f;
#pragma unroll 4
    for (int c = 0; c < 16; ++c) {
      v4f v = sp[c];
#pragma unroll
      for (int e = 0; e < 4; ++e) {
        const int j = jb0 + c * 4 + e;
        const float ej = (j <= t) ? __expf(v[e] * scale - m) : 0.f;
        v[e] = ej;
        sum += ej;
      }
      sp[c] = v;
    }
    red2[tid] = sum;
    __syncthreads();
    const float inv = 1.f / (red2[2 * t] + red2[2 * t + 1]);
    v4h* wp = (v4h*)(ws + t * XS + jb0);
#pragma unroll 4
    for (int c = 0; c < 16; ++c) {
      v4f v = sp[c];
      v4h pk;
#pragma unroll
      for (int e = 0; e < 4; ++e) pk[e] = (_Float16)(v[e] * inv);
      wp[c] = pk;
    }
  }
  __syncthreads();

  // --- out = weights . v  (B[s][h] = vt[h][s] -> vt rows are Brows) ------
  {
    float* ob = out + (size_t)b * SEQ_T * DIM_C;
    v16h a0 = frag_a(ws, m0, 0, lane);
    v16h a1 = frag_a(ws, m0, 32, lane);
    v16h a2 = frag_a(ws, m0, 64, lane);
    v16h a3 = frag_a(ws, m0, 96, lane);
    const int mrow = m0 + ((lane >> 4) << 3);
    for (int nt = 0; nt < 8; ++nt) {
      const int n0 = nt << 4;
      v8f acc = {};
      acc = wmma16(a0, frag_b_row(vt, n0, 0, lane), acc);
      acc = wmma16(a1, frag_b_row(vt, n0, 32, lane), acc);
      acc = wmma16(a2, frag_b_row(vt, n0, 64, lane), acc);
      acc = wmma16(a3, frag_b_row(vt, n0, 96, lane), acc);
      const int ncol = n0 + (lane & 15);
#pragma unroll
      for (int i = 0; i < 8; ++i)
        ob[(size_t)(mrow + i) * DIM_C + ncol] = acc[i];
    }
  }
}

extern "C" void kernel_launch(void* const* d_in, const int* in_sizes, int n_in,
                              void* d_out, int out_size, void* d_ws, size_t ws_size,
                              hipStream_t stream) {
  const float* x  = (const float*)d_in[0];
  const float* Wq = (const float*)d_in[1];
  const float* Wk = (const float*)d_in[2];
  const float* Wv = (const float*)d_in[3];
  float* out = (float*)d_out;
  _Float16* wf16 = (_Float16*)d_ws;                    // 98304 bytes used
  const int nb = in_sizes[0] / (SEQ_T * DIM_C);        // 2048 batch elements

  (void)n_in; (void)out_size; (void)ws_size;
  (void)hipFuncSetAttribute((const void*)attn128_kernel,
                            hipFuncAttributeMaxDynamicSharedMemorySize, LDS_BYTES);

  // 1) convert weights f32 -> f16 once (L2-resident thereafter)
  wcvt_kernel<<<192, 256, 0, stream>>>(Wq, Wk, Wv, wf16);
  // 2) fused attention, one workgroup (8 wave32) per batch element
  attn128_kernel<<<nb, 256, LDS_BYTES, stream>>>(x, wf16, out);
}
